// Attention_64862596104824
// MI455X (gfx1250) — matrix operand order repaired and checked by reading, not compiled, not run on hardware
//
#include <hip/hip_runtime.h>
#include <hip/hip_bf16.h>

typedef __attribute__((ext_vector_type(16))) _Float16 v16h;
typedef __attribute__((ext_vector_type(8)))  _Float16 v8h;
typedef __attribute__((ext_vector_type(8)))  float    v8f;
typedef __attribute__((ext_vector_type(4)))  unsigned int tdm_u4;
typedef __attribute__((ext_vector_type(8)))  int          tdm_i8;
typedef __attribute__((ext_vector_type(4)))  int          tdm_i4;

#define NB    4
#define NT    8192
#define NH    1024
#define NHEAD 16
#define DH    64
#define TS    8     // T-splits for the sequence-softmax / ctx reductions
#define PITCH  40   // halves per LDS row for K=32 f16 tiles (80B, 16B aligned)
#define PITCH2 72   // halves per LDS row for K=64 f16 tiles (144B, 16B aligned)

// ---------------------------------------------------------------------------
// Tensor Data Mover: 2-D tile (tile_x x tile_y elems, 4B elems) -> LDS.
// D# packing per CDNA5 ISA 8.3/8.4: group0 = {count=1, lds_addr, global_addr,
// type=2}; group1 = {data_size=4B, tensor dims, tile dims, row stride}.
// No LDS padding -> TDM packs rows contiguously (pitch = tile_x elems).
// Toolchain here exposes the 6-arg builtin (clang-23 / therock-10.0 lane).
// ---------------------------------------------------------------------------
__device__ __forceinline__ unsigned lds_off32(const void* p) {
  return (unsigned)(unsigned long long)p;   // low 32 bits = LDS byte offset
}

__device__ __forceinline__ void tdm_load_2d(unsigned lds_byte_off,
                                            const float* gptr,
                                            unsigned tile_x, unsigned tile_y,
                                            unsigned tensor_x, unsigned tensor_y,
                                            unsigned row_stride_elems)
{
  unsigned long long ga = (unsigned long long)gptr;
  tdm_u4 g0;
  g0[0] = 1u;                                          // count=1, user mode
  g0[1] = lds_byte_off;                                // lds_addr (bytes)
  g0[2] = (unsigned)ga;                                // global_addr[31:0]
  g0[3] = (unsigned)((ga >> 32) & 0x1ffffffu)          // global_addr[56:32]
          | (2u << 30);                                // type = 2 ("image")
  tdm_i8 g1;
  g1[0] = (int)(2u << 16);                             // data_size=4B; no flags
  g1[1] = (int)((tensor_x & 0xffffu) << 16);           // tensor_dim0[15:0]
  g1[2] = (int)((tensor_x >> 16) | ((tensor_y & 0xffffu) << 16));
  g1[3] = (int)((tensor_y >> 16) | (tile_x << 16));    // tile_dim0
  g1[4] = (int)tile_y;                                 // tile_dim1; tile_dim2=0
  g1[5] = (int)row_stride_elems;                       // dim0_stride[31:0]
  g1[6] = 0;                                           // stride hi / dim1_stride
  g1[7] = 0;
  tdm_i4 z4 = {0, 0, 0, 0};
  tdm_i8 z8 = {0, 0, 0, 0, 0, 0, 0, 0};
  __builtin_amdgcn_tensor_load_to_lds(g0, g1, z4, z4, z8, 0);
}

// ---- WMMA fragment loaders (contiguous k = kbase + s, same for A and B, wave32) ----------
__device__ __forceinline__ v16h load_a16(const _Float16* rowp, int kbase) {
  v8h lo = *(const v8h*)(rowp + kbase);
  v8h hi = *(const v8h*)(rowp + kbase + 8);
  v16h r;
#pragma unroll
  for (int i = 0; i < 8; ++i) { r[i] = lo[i]; r[i + 8] = hi[i]; }
  return r;
}
__device__ __forceinline__ v16h load_b16(const _Float16* rowp, int kbase) {
  v8h lo = *(const v8h*)(rowp + kbase);
  v8h hi = *(const v8h*)(rowp + kbase + 8);
  v16h r;
#pragma unroll
  for (int i = 0; i < 8; ++i) { r[i] = lo[i]; r[i + 8] = hi[i]; }
  return r;
}
// Same fragments but sourced from fp32 LDS tiles (TDM path), converting here.
__device__ __forceinline__ v16h frag_a_f32(const float* rowp, int kbase) {
  v8f lo = *(const v8f*)(rowp + kbase);
  v8f hi = *(const v8f*)(rowp + kbase + 8);
  v16h r;
#pragma unroll
  for (int i = 0; i < 8; ++i) { r[i] = (_Float16)lo[i]; r[i + 8] = (_Float16)hi[i]; }
  return r;
}
__device__ __forceinline__ v16h frag_b_f32(const float* rowp, int kbase) {
  v8f lo = *(const v8f*)(rowp + kbase);
  v8f hi = *(const v8f*)(rowp + kbase + 8);
  v16h r;
#pragma unroll
  for (int i = 0; i < 8; ++i) { r[i] = (_Float16)lo[i]; r[i + 8] = (_Float16)hi[i]; }
  return r;
}
__device__ __forceinline__ v8f wmma_f16(v16h a, v16h b, v8f c) {
  return __builtin_amdgcn_wmma_f32_16x16x32_f16(false, a, false, b,
                                                (short)0, c, false, false);
}

// ---------------------------------------------------------------------------
// Kernel A: Y[M,N] (fp16) = X[M,K] @ W[N,K]^T + bias.
// TDM double-buffered fp32 tiles in LDS; 8 waves, each a 32x32 subtile.
// Wave 0 drives the DMA: issue chunk i+1, wait TENSORcnt<=2 (chunk i done).
// ---------------------------------------------------------------------------
__global__ __launch_bounds__(256) void proj_gemm(
    const float* __restrict__ X, const float* __restrict__ W,
    const float* __restrict__ bias, _Float16* __restrict__ Y,
    int M, int N, int K)
{
  __shared__ __align__(32) float Xl[2][128 * 32];
  __shared__ __align__(32) float Wl[2][64 * 32];
  const int tid  = threadIdx.x;
  const int lane = tid & 31, wv = tid >> 5;
  const int lrow = lane & 15, hi16 = lane >> 4;
  const int m0 = blockIdx.y * 128, n0 = blockIdx.x * 64;
  const int wm = (wv >> 1) * 32, wn = (wv & 1) * 32;
  v8f acc[2][2] = {};
  const int nch = K >> 5;

  if (wv == 0) {
    tdm_load_2d(lds_off32(&Xl[0][0]), X + (size_t)m0 * K, 32, 128,
                (unsigned)K, (unsigned)M, (unsigned)K);
    tdm_load_2d(lds_off32(&Wl[0][0]), W + (size_t)n0 * K, 32, 64,
                (unsigned)K, (unsigned)N, (unsigned)K);
  }

  for (int ic = 0; ic < nch; ++ic) {
    const int cur = ic & 1;
    if (wv == 0) {
      if (ic + 1 < nch) {
        const int k1 = (ic + 1) << 5;
        tdm_load_2d(lds_off32(&Xl[cur ^ 1][0]), X + (size_t)m0 * K + k1,
                    32, 128, (unsigned)K, (unsigned)M, (unsigned)K);
        tdm_load_2d(lds_off32(&Wl[cur ^ 1][0]), W + (size_t)n0 * K + k1,
                    32, 64, (unsigned)K, (unsigned)N, (unsigned)K);
        __builtin_amdgcn_s_wait_tensorcnt(2);   // current chunk landed
      } else {
        __builtin_amdgcn_s_wait_tensorcnt(0);
      }
    }
    __syncthreads();                            // tiles visible to all waves

    const float* Xb = &Xl[cur][0];
    const float* Wb = &Wl[cur][0];
    v16h a[2], bm[2];
#pragma unroll
    for (int mi = 0; mi < 2; ++mi)
      a[mi] = frag_a_f32(Xb + (wm + mi * 16 + lrow) * 32, hi16 * 16);
#pragma unroll
    for (int ni = 0; ni < 2; ++ni)
      bm[ni] = frag_b_f32(Wb + (wn + ni * 16 + lrow) * 32, hi16 * 16);
#pragma unroll
    for (int mi = 0; mi < 2; ++mi)
#pragma unroll
      for (int ni = 0; ni < 2; ++ni)
        acc[mi][ni] = wmma_f16(a[mi], bm[ni], acc[mi][ni]);

    __syncthreads();                            // done reading before reuse
  }

#pragma unroll
  for (int ni = 0; ni < 2; ++ni) {
    const int col = n0 + wn + ni * 16 + lrow;
    const float bv = bias[col];
#pragma unroll
    for (int mi = 0; mi < 2; ++mi)
#pragma unroll
      for (int r = 0; r < 8; ++r) {
        int row = m0 + wm + mi * 16 + r + hi16 * 8;
        Y[(size_t)row * N + col] = (_Float16)(acc[mi][ni][r] + bv);
      }
  }
}

// ---------------------------------------------------------------------------
// Kernel B1a/B1b: split column softmax stats over T (8-way) + merge.
// ---------------------------------------------------------------------------
__global__ __launch_bounds__(256) void col_stats_part(
    const _Float16* __restrict__ kp, const float* __restrict__ mask_attn,
    float* __restrict__ pm, float* __restrict__ ps)
{
  const int c = blockIdx.x * 256 + threadIdx.x;
  const int b = blockIdx.y;
  const int ts = blockIdx.z;
  const int tBeg = ts * (NT / TS), tEnd = tBeg + (NT / TS);
  const _Float16* base = kp + (size_t)b * NT * NH + c;
  float m = -1e30f, s = 0.f;
  for (int t = tBeg; t < tEnd; ++t) {
    float x = (float)base[(size_t)t * NH] - 10000.f * (1.f - mask_attn[b * NT + t]);
    float mn = fmaxf(m, x);
    s = s * __expf(m - mn) + __expf(x - mn);
    m = mn;
  }
  pm[((size_t)ts * NB + b) * NH + c] = m;
  ps[((size_t)ts * NB + b) * NH + c] = s;
}

__global__ __launch_bounds__(256) void col_stats_merge(
    const float* __restrict__ pm, const float* __restrict__ ps,
    float* __restrict__ colmax, float* __restrict__ colsum)
{
  const int c = blockIdx.x * 256 + threadIdx.x;
  const int b = blockIdx.y;
  float m = -1e30f, s = 0.f;
#pragma unroll
  for (int ts = 0; ts < TS; ++ts) {
    float mm = pm[((size_t)ts * NB + b) * NH + c];
    float ss = ps[((size_t)ts * NB + b) * NH + c];
    float mn = fmaxf(m, mm);
    s = s * __expf(m - mn) + ss * __expf(mm - mn);
    m = mn;
  }
  colmax[b * NH + c] = m;
  colsum[b * NH + c] = s;
}

__global__ __launch_bounds__(256) void zero_f32(float* __restrict__ p, int n) {
  int i = blockIdx.x * 256 + threadIdx.x;
  if (i < n) p[i] = 0.f;
}

// ---------------------------------------------------------------------------
// Kernel B2: ctx[b,h,d,e] += (1/8) * sum_{t in split} softmaxK[t,d] * v[t,e]
// grid = (b*h, T-split); partial sums combined with global_atomic_add_f32.
// ---------------------------------------------------------------------------
__global__ __launch_bounds__(256) void ctx_kernel(
    const _Float16* __restrict__ kp, const _Float16* __restrict__ vp,
    const float* __restrict__ mask_attn,
    const float* __restrict__ colmax, const float* __restrict__ colsum,
    float* __restrict__ ctx)
{
  __shared__ __align__(16) _Float16 At[64 * PITCH];  // [d][t]
  __shared__ __align__(16) _Float16 Bt[64 * PITCH];  // [e][t]
  __shared__ float cm[64], rs[64];

  const int bh = blockIdx.x;
  const int b = bh >> 4, h = bh & 15;
  const int ts = blockIdx.y;
  const int tBeg = ts * (NT / TS), tEnd = tBeg + (NT / TS);
  const int tid = threadIdx.x, lane = tid & 31, wv = tid >> 5;
  const int lrow = lane & 15, hi16 = lane >> 4;
  const int dt = wv >> 1, et0 = (wv & 1) * 2;

  if (tid < 64) {
    cm[tid] = colmax[b * NH + h * DH + tid];
    rs[tid] = 1.f / colsum[b * NH + h * DH + tid];
  }

  const int tt = tid >> 3;            // 0..31 : t within chunk
  const int d0 = (tid & 7) * 8;       // 0..56 : 8 features per thread
  v8f acc[2] = {};

  for (int t0 = tBeg; t0 < tEnd; t0 += 32) {
    __syncthreads();
    const size_t rowoff = ((size_t)(b * NT + t0 + tt)) * NH + h * DH + d0;
    const float ma = 10000.f * (1.f - mask_attn[b * NT + t0 + tt]);
    v8h kv = *(const v8h*)(kp + rowoff);
    v8h vv = *(const v8h*)(vp + rowoff);
#pragma unroll
    for (int j = 0; j < 8; ++j) {
      int d = d0 + j;
      float p = __expf((float)kv[j] - ma - cm[d]) * rs[d];
      At[d * PITCH + tt] = (_Float16)p;
      Bt[d * PITCH + tt] = vv[j];
    }
    __syncthreads();

    v16h a  = load_a16(&At[(dt * 16 + lrow) * PITCH], hi16 * 16);
    v16h b0 = load_b16(&Bt[((et0 + 0) * 16 + lrow) * PITCH], hi16 * 16);
    v16h b1 = load_b16(&Bt[((et0 + 1) * 16 + lrow) * PITCH], hi16 * 16);
    acc[0] = wmma_f16(a, b0, acc[0]);
    acc[1] = wmma_f16(a, b1, acc[1]);
  }

#pragma unroll
  for (int e2 = 0; e2 < 2; ++e2)
#pragma unroll
    for (int r = 0; r < 8; ++r) {
      int d = dt * 16 + r + hi16 * 8;
      int e = (et0 + e2) * 16 + lrow;
      atomicAdd(&ctx[(((size_t)b * NHEAD + h) * DH + d) * DH + e],
                acc[e2][r] * 0.125f);
    }
}

// ---------------------------------------------------------------------------
// Kernel C: out[b,t,h*64+e] = sum_d softmax_d(qh[b,h,t,:])[d] * ctx[b,h,d,e]
// ---------------------------------------------------------------------------
__global__ __launch_bounds__(256) void out_kernel(
    const _Float16* __restrict__ qp, const float* __restrict__ ctx,
    const float* __restrict__ mask_q, float* __restrict__ out)
{
  __shared__ __align__(16) _Float16 S[64 * PITCH2];   // [t][d]
  __shared__ __align__(16) _Float16 Ct[64 * PITCH2];  // [e][d]

  const int t0 = blockIdx.x * 64;
  const int h = blockIdx.y, b = blockIdx.z;
  const int tid = threadIdx.x, lane = tid & 31, wv = tid >> 5;
  const int lrow = lane & 15, hi16 = lane >> 4;
  const int tt = wv >> 1, et0 = (wv & 1) * 2;

#pragma unroll
  for (int i = 0; i < 16; ++i) {
    int idx = tid + i * 256;
    int d = idx >> 6, e = idx & 63;
    Ct[e * PITCH2 + d] =
        (_Float16)ctx[(((size_t)b * NHEAD + h) * DH + d) * DH + e];
  }

  if (tid < 64) {
    const int t = t0 + tid;
    const _Float16* qrow = qp + ((size_t)(b * NT + t)) * NH + h * DH;
    const float mq = 10000.f * (1.f - mask_q[b * NT + t]);
    float m = -1e30f;
    for (int d = 0; d < DH; ++d) m = fmaxf(m, (float)qrow[d] - mq);
    float s = 0.f;
    for (int d = 0; d < DH; ++d) s += __expf((float)qrow[d] - mq - m);
    const float r = 1.f / s;
    for (int d = 0; d < DH; ++d)
      S[tid * PITCH2 + d] = (_Float16)(__expf((float)qrow[d] - mq - m) * r);
  }
  __syncthreads();

  v8f acc[2] = {};
#pragma unroll
  for (int kc = 0; kc < 2; ++kc) {
    v16h a  = load_a16(&S[(tt * 16 + lrow) * PITCH2], kc * 32 + hi16 * 16);
    v16h b0 = load_b16(&Ct[((et0 + 0) * 16 + lrow) * PITCH2], kc * 32 + hi16 * 16);
    v16h b1 = load_b16(&Ct[((et0 + 1) * 16 + lrow) * PITCH2], kc * 32 + hi16 * 16);
    acc[0] = wmma_f16(a, b0, acc[0]);
    acc[1] = wmma_f16(a, b1, acc[1]);
  }

#pragma unroll
  for (int e2 = 0; e2 < 2; ++e2)
#pragma unroll
    for (int r = 0; r < 8; ++r) {
      int t = t0 + tt * 16 + r + hi16 * 8;
      int e = (et0 + e2) * 16 + lrow;
      out[((size_t)(b * NT + t)) * NH + h * DH + e] = acc[e2][r];
    }
}

// ---------------------------------------------------------------------------
extern "C" void kernel_launch(void* const* d_in, const int* in_sizes, int n_in,
                              void* d_out, int out_size, void* d_ws, size_t ws_size,
                              hipStream_t stream) {
  const float* q         = (const float*)d_in[0];
  const float* k         = (const float*)d_in[1];
  const float* v         = (const float*)d_in[2];
  const float* mask_q    = (const float*)d_in[3];
  const float* mask_attn = (const float*)d_in[4];
  const float* Wq        = (const float*)d_in[5];
  const float* bq        = (const float*)d_in[6];
  const float* Wk        = (const float*)d_in[7];
  const float* bk        = (const float*)d_in[8];
  const float* Wv        = (const float*)d_in[9];
  const float* bv        = (const float*)d_in[10];

  const size_t M = (size_t)NB * NT;          // 32768
  const size_t P = M * NH;                   // elements per projection
  char* ws = (char*)d_ws;
  _Float16* qp = (_Float16*)(ws);
  _Float16* kp = (_Float16*)(ws + P * 2);
  _Float16* vp = (_Float16*)(ws + 2 * P * 2);
  float* colmax = (float*)(ws + 3 * P * 2);
  float* colsum = colmax + NB * NH;
  float* ctx    = colsum + NB * NH;          // NB*NHEAD*DH*DH floats
  float* pm     = ctx + (size_t)NB * NHEAD * DH * DH;
  float* ps     = pm + (size_t)TS * NB * NH;

  const int ctxN = NB * NHEAD * DH * DH;
  zero_f32<<<dim3((ctxN + 255) / 256), 256, 0, stream>>>(ctx, ctxN);

  dim3 gGemm(NH / 64, (unsigned)(M / 128), 1);
  proj_gemm<<<gGemm, 256, 0, stream>>>(q, Wq, bq, qp, (int)M, NH, NH);
  proj_gemm<<<gGemm, 256, 0, stream>>>(k, Wk, bk, kp, (int)M, NH, NH);
  proj_gemm<<<gGemm, 256, 0, stream>>>(v, Wv, bv, vp, (int)M, NH, NH);

  col_stats_part<<<dim3(NH / 256, NB, TS), 256, 0, stream>>>(kp, mask_attn, pm, ps);
  col_stats_merge<<<dim3(NH / 256, NB), 256, 0, stream>>>(pm, ps, colmax, colsum);

  ctx_kernel<<<dim3(NB * NHEAD, TS), 256, 0, stream>>>(kp, vp, mask_attn,
                                                       colmax, colsum, ctx);

  out_kernel<<<dim3(NT / 64, NHEAD, NB), 256, 0, stream>>>(qp, ctx, mask_q,
                                                           (float*)d_out);
}
